// DPModel_47218870452459
// MI455X (gfx1250) — matrix operand order repaired and checked by reading, not compiled, not run on hardware
//
#include <hip/hip_runtime.h>
#include <math.h>

// ---------------- problem constants (from reference) ----------------
#define NATOM   4096
#define MNBR    128
#define CDIM    64
#define XDIM    10
#define AXISD   16
#define HALFN   2048      // TYPE_IDX[1]
#define RCUT_F  6.0f
#define NORM_F  20.0f

typedef __attribute__((ext_vector_type(16))) _Float16 v16h;
typedef __attribute__((ext_vector_type(8)))  _Float16 v8h;
typedef __attribute__((ext_vector_type(8)))  float    v8f;
typedef __attribute__((ext_vector_type(4)))  unsigned int v4u;
typedef __attribute__((ext_vector_type(8)))  int      v8i;
typedef __attribute__((ext_vector_type(4)))  int      v4i;

// CDNA5 has native v_tanh_f32 (TANH is a listed transcendental VALU op).
#if __has_builtin(__builtin_amdgcn_tanhf)
#warning "CDNA5 probe: native v_tanh_f32 path ENABLED"
__device__ __forceinline__ float fast_tanh(float x) {
  return __builtin_amdgcn_tanhf(x);
}
#else
#warning "CDNA5 probe: native tanh NOT available - exp/rcp fallback"
// Branch-free tanh: 1 - 2/(exp(2x)+1)  (v_exp_f32 + v_rcp_f32)
__device__ __forceinline__ float fast_tanh(float x) {
  float e = __expf(2.0f * x);
  return 1.0f - 2.0f * __builtin_amdgcn_rcpf(e + 1.0f);
}
#endif

// ---------------- WMMA fragment loaders (ISA 7.12.2 layouts, wave32) ----
// A (16x32 f16) from row-major tile: two contiguous 16B runs per lane.
__device__ __forceinline__ v16h load_a_frag_vec(const _Float16* __restrict__ tile,
                                                int ldm, int lane) {
  int row  = lane & 15;
  int koff = (lane < 16) ? 0 : 8;
  const _Float16* p = tile + row * ldm + koff;
  v8h lo = *(const v8h*)(p);
  v8h hi = *(const v8h*)(p + 16);
  v16h a;
#pragma unroll
  for (int h = 0; h < 8; ++h) { a[h] = lo[h]; a[8 + h] = hi[h]; }
  return a;
}

// A fragment where only K=0..15 is valid (upper half zero): one b128 load.
__device__ __forceinline__ v16h load_a_frag_k16(const _Float16* __restrict__ tile,
                                                int ldm, int lane) {
  int row  = lane & 15;
  int koff = (lane < 16) ? 0 : 8;
  v8h lo = *(const v8h*)(tile + row * ldm + koff);
  v16h a;
#pragma unroll
  for (int h = 0; h < 8; ++h) { a[h] = lo[h]; a[8 + h] = (_Float16)0.0f; }
  return a;
}

// B fragment from a TRANSPOSED [n][k] tile: each lane's 16 K-halves are two
// 8-half runs (k0+kb, k0+kb+16) -> two b128 loads. kvalid masks the K window (compile-time).
__device__ __forceinline__ v16h load_b_frag_tr(const _Float16* __restrict__ tileT,
                                               int ldk, int col0, int k0,
                                               int kvalid, int lane) {
  int col = (lane & 15) + col0;
  int kb  = (lane < 16) ? 0 : 8;
  const _Float16* p = tileT + col * ldk + k0 + kb;
  v8h lo = *(const v8h*)(p);
  v8h hi = *(const v8h*)(p + 16);
  v16h b;
#pragma unroll
  for (int h = 0; h < 8; ++h) {
    b[h]     = (kb + h     < kvalid) ? lo[h] : (_Float16)0.0f;
    b[8 + h] = (kb + 16 + h < kvalid) ? hi[h] : (_Float16)0.0f;
  }
  return b;
}

// B fragment converted on the fly from row-major f32 weights (small nets).
__device__ __forceinline__ v16h load_b_frag_f32(const float* __restrict__ w,
                                                int ldn, int col0, int kvalid,
                                                int lane) {
  int col = (lane & 15) + col0;
  int kb  = (lane < 16) ? 0 : 8;
  v16h b;
#pragma unroll
  for (int h = 0; h < 16; ++h) {
    int k = (h < 8) ? (kb + h) : (16 + kb + (h - 8));
    b[h] = (k < kvalid) ? (_Float16)w[k * ldn + col] : (_Float16)0.0f;
  }
  return b;
}

__device__ __forceinline__ v8f wmma_f16(v16h a, v16h b, v8f c) {
  return __builtin_amdgcn_wmma_f32_16x16x32_f16(false, a, false, b,
                                                (short)0, c, false, false);
}

// ---------------- Tensor Data Mover staging (gfx1250 TDM) ---------------
#if __has_builtin(__builtin_amdgcn_tensor_load_to_lds) && \
    __has_builtin(__builtin_amdgcn_s_wait_tensorcnt)
#define TDM_AVAILABLE 1
#warning "CDNA5 probe: TDM builtin path ENABLED (tensor_load_to_lds)"
#else
#define TDM_AVAILABLE 0
#warning "CDNA5 probe: TDM builtin NOT available - cooperative-copy fallback"
#endif

#if TDM_AVAILABLE
// 2D f16 tile load: tile1 rows x tile0 elems, row stride in elements.
__device__ __forceinline__ void tdm_load_2d_f16(const _Float16* gsrc,
                                                unsigned lds_off,
                                                unsigned tensor_d0,
                                                unsigned tensor_d1,
                                                unsigned tile0, unsigned tile1,
                                                unsigned row_stride) {
  unsigned long long ga = (unsigned long long)(size_t)gsrc;
  v4u g0;
  g0[0] = 1u;                                         // count=1, user mode
  g0[1] = lds_off;                                    // lds_addr
  g0[2] = (unsigned)(ga & 0xFFFFFFFFu);               // global_addr[31:0]
  g0[3] = (unsigned)((ga >> 32) & 0x1FFFFFFu) | (2u << 30);  // [56:32], type=2
  v8i g1;
  g1[0] = (int)(1u << 16);                            // data_size = 2 bytes
  g1[1] = (int)((tensor_d0 & 0xFFFFu) << 16);         // tensor_dim0 lo
  g1[2] = (int)((tensor_d0 >> 16) | ((tensor_d1 & 0xFFFFu) << 16));
  g1[3] = (int)((tensor_d1 >> 16) | ((tile0 & 0xFFFFu) << 16));
  g1[4] = (int)(tile1 & 0xFFFFu);                     // tile_dim1; tile_dim2=0
  g1[5] = (int)row_stride;                            // tensor_dim0_stride lo32
  g1[6] = 0;
  g1[7] = 0;
  v4i z4 = {0, 0, 0, 0};
#if __clang_major__ >= 23
  v8i z8 = {0, 0, 0, 0, 0, 0, 0, 0};
  __builtin_amdgcn_tensor_load_to_lds(g0, g1, z4, z4, z8, 0);
#else
  __builtin_amdgcn_tensor_load_to_lds(g0, g1, z4, z4, 0);
#endif
}
#endif

// ---------------- K1: geometry -> srcn (f16), R channels (f16, padded) ---
__global__ __launch_bounds__(256) void geom_kernel(
    const float* __restrict__ coord, const float* __restrict__ box,
    const int* __restrict__ nbrs,
    _Float16* __restrict__ srcn_out,        // [N*M]
    _Float16* __restrict__ R_out) {         // [N][16][M]  (x-major, padded)
  int idx = blockIdx.x * 256 + threadIdx.x;
  if (idx >= NATOM * MNBR) return;
  int n = idx >> 7;
  int m = idx & 127;
  int j = nbrs[idx];
  float x[3];
#pragma unroll
  for (int d = 0; d < 3; ++d) {
    float b = box[d];
    float rb = __builtin_amdgcn_rcpf(b);
    float dx = coord[d * NATOM + j] - coord[d * NATOM + n];
    dx -= b * rintf(dx * rb);
    x[d] = dx;
  }
  float r2 = x[0]*x[0] + x[1]*x[1] + x[2]*x[2];
  bool mask = r2 > 1e-12f;
  float rsafe = sqrtf(mask ? r2 : 1.0f);
  float r = mask ? rsafe : 0.0f;
  // native v_cos_f32 switching function
  float sw = 0.5f * __cosf(3.14159265358979323846f * fminf(r, RCUT_F) *
                           (1.0f / RCUT_F)) + 0.5f;
  float rinv = __builtin_amdgcn_rcpf(rsafe);
  float sr = (mask && r < RCUT_F) ? sw * rinv : 0.0f;
  const float mean[2] = {0.1f, 0.1f};
  const float stdv[2] = {0.05f, 0.05f};
  int ti = (n >= HALFN) ? 1 : 0;
  float srn  = sr / stdv[ti];
  float srcn = (sr - mean[ti]) / stdv[ti];
  float inv = mask ? rinv : 1.0f;
  float xn[3] = {x[0]*inv, x[1]*inv, x[2]*inv};
  float R[XDIM];
  R[0] = srn;
  const float s3 = 1.7320508075688772f, s18 = 4.242640687119285f;
#pragma unroll
  for (int d = 0; d < 3; ++d) {
    R[1 + d] = s3 * srn * xn[d];
    R[4 + d] = 3.0f * srn * (xn[d]*xn[d] - (1.0f/3.0f));
    R[7 + d] = s18 * srn * (xn[d] * xn[(d + 1) % 3]);
  }
  srcn_out[idx] = (_Float16)srcn;
  _Float16* Rn = R_out + (size_t)n * 16 * MNBR;
#pragma unroll
  for (int xch = 0; xch < XDIM; ++xch)
    Rn[xch * MNBR + m] = (_Float16)R[xch];
  // threads m=122..127 zero pad rows 10..15 (vectorized b128 stores)
  if (m >= 122) {
    int xpad = 10 + (m - 122);
    v8h z;
#pragma unroll
    for (int h = 0; h < 8; ++h) z[h] = (_Float16)0.0f;
    _Float16* pr = Rn + xpad * MNBR;
    for (int c = 0; c < MNBR; c += 8) *(v8h*)(pr + c) = z;
  }
}

// ---------------- K2: fused embedding MLP + T einsum (1 wave / atom) -----
__global__ __launch_bounds__(32) void embed_T_kernel(
    const _Float16* __restrict__ srcn, const _Float16* __restrict__ Rh,
    const float* __restrict__ eW0, const float* __restrict__ eb0,
    const float* __restrict__ eW1, const float* __restrict__ eb1,
    const float* __restrict__ eW2, const float* __restrict__ eb2,
    const float* __restrict__ Tbias,
    float* __restrict__ T_out) {            // [N][10][64]
  __shared__ __align__(16) _Float16 sA[16 * 32];   // layer-1 A tile (K 16->32)
  __shared__ __align__(16) _Float16 sH[16 * 32];   // layer-2 A tile (K=32)
  __shared__ __align__(16) _Float16 sEt[16 * 16];  // embed tile TRANSPOSED [c][m]
  int n = blockIdx.x;
  int lane = threadIdx.x;
  int ti = (n >= HALFN) ? 1 : 0;
  int ncol = lane & 15;
  int mbase = (lane < 16) ? 0 : 8;
  const _Float16* Rn = Rh + (size_t)n * 16 * MNBR;

  v8f Tacc[4] = {};
  for (int z = lane; z < 16 * 32; z += 32) sA[z] = (_Float16)0.0f;
  __syncthreads();

  int row = lane & 15;
  int c0  = (lane >> 4) * 8;

  for (int jb = 0; jb < 2; ++jb) {
    int k = ti * 2 + jb;                     // embed-net index i*NTYPES+j
    // hoist the (tiny) weight fragments and biases for this net into regs
    float w0c[8], b0c[8];
#pragma unroll
    for (int c = 0; c < 8; ++c) {
      w0c[c] = eW0[k * 16 + c0 + c];
      b0c[c] = eb0[k * 16 + c0 + c];
    }
    v16h b1f[2];
    float bb1[2];
#pragma unroll
    for (int nt = 0; nt < 2; ++nt) {
      b1f[nt] = load_b_frag_f32(eW1 + k * 16 * 32, 32, nt * 16, 16, lane);
      bb1[nt] = eb1[k * 32 + nt * 16 + ncol];
    }
    v16h b2f[4];
    float bb2[4];
#pragma unroll
    for (int ct = 0; ct < 4; ++ct) {
      b2f[ct] = load_b_frag_f32(eW2 + k * 32 * 64, 64, ct * 16, 32, lane);
      bb2[ct] = eb2[k * 64 + ct * 16 + ncol];
    }

    for (int t = jb * 4; t < jb * 4 + 4; ++t) {
      int m0 = t * 16;
      // ---- layer 0: scalar -> 16, tanh, into A tile (cols 0..15)
      float xin = (float)srcn[n * MNBR + m0 + row];
      __syncthreads();
#pragma unroll
      for (int c = 0; c < 8; ++c)
        sA[row * 32 + c0 + c] = (_Float16)fast_tanh(xin * w0c[c] + b0c[c]);
      __syncthreads();
      v16h a1 = load_a_frag_vec(sA, 32, lane);
      // ---- layer 1: (16x16)@(16x32), K padded to 32 -> h1 in sH
#pragma unroll
      for (int nt = 0; nt < 2; ++nt) {
        v8f c1 = {};
        c1 = wmma_f16(a1, b1f[nt], c1);
#pragma unroll
        for (int v = 0; v < 8; ++v)
          sH[(v + mbase) * 32 + nt * 16 + ncol] =
              (_Float16)fast_tanh(c1[v] + bb1[nt]);
      }
      __syncthreads();
      v16h a2 = load_a_frag_vec(sH, 32, lane);
      // ---- layer 2 (+ tanh) and T accumulation per 16-col c-tile
#pragma unroll
      for (int ct = 0; ct < 4; ++ct) {
        v8f e = {};
        e = wmma_f16(a2, b2f[ct], e);
        __syncthreads();
#pragma unroll
        for (int v = 0; v < 8; ++v)   // transposed store: sEt[c][m]
          sEt[ncol * 16 + (v + mbase)] = (_Float16)fast_tanh(e[v] + bb2[ct]);
        __syncthreads();
        v16h aR = load_a_frag_k16(Rn + m0, MNBR, lane);
        v16h bE = load_b_frag_tr(sEt, 16, 0, 0, 16, lane);
        Tacc[ct] = wmma_f16(aR, bE, Tacc[ct]);
      }
      __syncthreads();
    }
  }
  // ---- write T (/= NORM, + Tbias on x==0 row)
#pragma unroll
  for (int ct = 0; ct < 4; ++ct) {
#pragma unroll
    for (int v = 0; v < 8; ++v) {
      int x = v + mbase;
      if (x < XDIM) {
        int c = ct * 16 + ncol;
        float val = Tacc[ct][v] / NORM_F + ((x == 0) ? Tbias[c] : 0.0f);
        T_out[(size_t)n * XDIM * CDIM + x * CDIM + c] = val;
      }
    }
  }
}

// ---------------- K3: G_NAC = U^T(16x10) @ V(10x64) via WMMA -------------
__global__ __launch_bounds__(32) void g_kernel(const float* __restrict__ T,
                                               _Float16* __restrict__ Gh) {
  __shared__ __align__(16) _Float16 sU[16 * 32];   // A tile [a][x], pad->32
  __shared__ __align__(16) _Float16 sVt[64 * 16];  // B TRANSPOSED [c][x]
  int n = blockIdx.x;
  int lane = threadIdx.x;
  const float* Tn = T + (size_t)n * XDIM * CDIM;
  for (int z = lane; z < 16 * 32; z += 32) sU[z] = (_Float16)0.0f;
  for (int z = lane; z < 64 * 16; z += 32) sVt[z] = (_Float16)0.0f;
  __syncthreads();
  for (int z = lane; z < XDIM * 64; z += 32) {
    int x = z >> 6, c = z & 63;
    sVt[c * 16 + x] = (_Float16)Tn[x * CDIM + c];
  }
  for (int z = lane; z < 16 * XDIM; z += 32) {
    int a = z & 15, x = z >> 4;
    float u;
    if (x == 0)      u = Tn[a];
    else if (x < 4)  u = Tn[x * CDIM + a];
    else {
      int xx = x - 4;
      float g11;
      if (xx < 3) {
        float g1 = Tn[(1 + xx) * CDIM + AXISD + a];
        g11 = g1 * g1;
      } else {
        int d = xx - 3, rd = (d + 1) % 3;
        float g1  = Tn[(1 + d)  * CDIM + AXISD + a];
        float g1r = Tn[(1 + rd) * CDIM + AXISD + a];
        g11 = 1.4142135623730951f * g1 * g1r;
      }
      u = g11 + Tn[(4 + xx) * CDIM + AXISD + a];
    }
    sU[a * 32 + x] = (_Float16)u;
  }
  __syncthreads();
  v16h a = load_a_frag_vec(sU, 32, lane);
  int ncol = lane & 15;
  int mbase = (lane < 16) ? 0 : 8;
#pragma unroll
  for (int ct = 0; ct < 4; ++ct) {
    v16h b = load_b_frag_tr(sVt, 16, ct * 16, 0, XDIM, lane);
    v8f g = {};
    g = wmma_f16(a, b, g);
#pragma unroll
    for (int v = 0; v < 8; ++v)
      Gh[(size_t)n * (AXISD * CDIM) + (v + mbase) * CDIM + ct * 16 + ncol] =
          (_Float16)g[v];
  }
}

// ---------------- K3b: convert + TRANSPOSE fitting weights to f16 --------
// W0t: [type][n=128][k=1024], W1t: [type][n=128][k=128]
__global__ __launch_bounds__(256) void cvt_kernel(
    const float* __restrict__ fW0, const float* __restrict__ fW1,
    _Float16* __restrict__ W0t, _Float16* __restrict__ W1t) {
  int i = blockIdx.x * 256 + threadIdx.x;
  if (i < 2 * 1024 * 128) {
    int t = i >> 17, rem = i & 131071;
    int k = rem >> 7, nn = rem & 127;
    W0t[((size_t)(t * 128 + nn)) * 1024 + k] = (_Float16)fW0[i];
  }
  if (i < 2 * 128 * 128) {
    int t = i >> 14, rem = i & 16383;
    int k = rem >> 7, nn = rem & 127;
    W1t[((size_t)(t * 128 + nn)) * 128 + k] = (_Float16)fW1[i];
  }
}

// ---------------- K4: fitting net, 4 waves / WG, TDM-staged B tiles ------
__global__ __launch_bounds__(128) void fit_kernel(
    const _Float16* __restrict__ Gh,
    const _Float16* __restrict__ W0t, const _Float16* __restrict__ W1t,
    const float* __restrict__ fb0, const float* __restrict__ fb1,
    const float* __restrict__ fW2, const float* __restrict__ fb2,
    const float* __restrict__ Ebias, float* __restrict__ out) {
  __shared__ __align__(16) _Float16 sBt[128 * 32];    // W0 chunk [n][k]  8 KB
  __shared__ __align__(16) _Float16 sB1t[128 * 128];  // W1 full  [n][k] 32 KB
  __shared__ __align__(16) _Float16 sH0[64 * 128];    // h0 tile         16 KB
  __shared__ float sRed[128];
  int t    = blockIdx.x >> 5;          // type
  int rg   = blockIdx.x & 31;          // 64-row group
  int wv   = threadIdx.x >> 5;
  int lane = threadIdx.x & 31;
  int r0   = t * HALFN + rg * 64 + wv * 16;
  int ncol = lane & 15, mbase = (lane < 16) ? 0 : 8;
  const _Float16* Arows = Gh + (size_t)r0 * 1024;
  const _Float16* W0 = W0t + (size_t)t * 128 * 1024;
  const _Float16* W1 = W1t + (size_t)t * 128 * 128;

  // stage the whole transposed W1 (128x128 f16) once via TDM; waited below
#if TDM_AVAILABLE
  if (threadIdx.x == 0)
    tdm_load_2d_f16(W1, (unsigned)(unsigned long long)(void*)sB1t,
                    128, 128, 128, 128, 128);
#else
  for (int z = threadIdx.x; z < 128 * 128; z += 128) sB1t[z] = W1[z];
#endif

  // ---- layer 0: (16 x 1024) @ (1024 x 128), K-chunks of 32 staged in LDS
  v8f acc[8] = {};
  for (int kk = 0; kk < 32; ++kk) {
    __syncthreads();   // previous chunk consumed
#if TDM_AVAILABLE
    if (threadIdx.x == 0) {
      // rows n=0..127 of W0t, columns kk*32..kk*32+31 (contiguous in k)
      tdm_load_2d_f16(W0 + (size_t)kk * 32,
                      (unsigned)(unsigned long long)(void*)sBt,
                      32, 128, 32, 128, 1024);
      __builtin_amdgcn_s_wait_tensorcnt(0);
    }
#else
    for (int z = threadIdx.x; z < 128 * 32; z += 128) {
      int nn = z >> 5, k = z & 31;
      sBt[z] = W0[(size_t)nn * 1024 + kk * 32 + k];
    }
#endif
    __syncthreads();   // chunk ready
    v16h a = load_a_frag_vec(Arows + kk * 32, 1024, lane);
#pragma unroll
    for (int ct = 0; ct < 8; ++ct) {
      v16h b = load_b_frag_tr(sBt, 32, ct * 16, 0, 32, lane);
      acc[ct] = wmma_f16(a, b, acc[ct]);
    }
  }
  __syncthreads();
#pragma unroll
  for (int ct = 0; ct < 8; ++ct) {
    float bias = fb0[t * 128 + ct * 16 + ncol];
#pragma unroll
    for (int v = 0; v < 8; ++v)
      sH0[(wv * 16 + v + mbase) * 128 + ct * 16 + ncol] =
          (_Float16)fast_tanh(acc[ct][v] + bias);
  }
  __syncthreads();

  // ---- layer 1: (16 x 128) @ (128 x 128), transposed B resident in LDS
  v8f acc1[8] = {};
#pragma unroll
  for (int kk = 0; kk < 4; ++kk) {
    v16h a = load_a_frag_vec(sH0 + (wv * 16) * 128 + kk * 32, 128, lane);
#pragma unroll
    for (int ct = 0; ct < 8; ++ct) {
      v16h b = load_b_frag_tr(sB1t, 128, ct * 16, kk * 32, 32, lane);
      acc1[ct] = wmma_f16(a, b, acc1[ct]);
    }
  }

  // ---- layer 2 folded into the fragments: energy += sum_r,c h1[r,c]*W2[c]
  float p = 0.0f;
#pragma unroll
  for (int ct = 0; ct < 8; ++ct) {
    float bias = fb1[t * 128 + ct * 16 + ncol];
    float w2   = fW2[t * 128 + ct * 16 + ncol];
#pragma unroll
    for (int v = 0; v < 8; ++v)
      p += fast_tanh(acc1[ct][v] + bias) * w2;
  }
  sRed[threadIdx.x] = p;
  __syncthreads();
  if (threadIdx.x == 0) {
    float s = 64.0f * (fb2[t] + Ebias[t]);   // 64 rows per block
    for (int i = 0; i < 128; ++i) s += sRed[i];
    atomicAdd(out, s);
  }
}

__global__ void zero_out_kernel(float* __restrict__ out) { out[0] = 0.0f; }

// ---------------- launch ----------------
extern "C" void kernel_launch(void* const* d_in, const int* in_sizes, int n_in,
                              void* d_out, int out_size, void* d_ws, size_t ws_size,
                              hipStream_t stream) {
  (void)in_sizes; (void)n_in; (void)out_size; (void)ws_size;
  const float* coord = (const float*)d_in[0];
  const float* box   = (const float*)d_in[1];
  const int*   nbrs  = (const int*)d_in[2];
  const float* eW0   = (const float*)d_in[3];
  const float* eb0   = (const float*)d_in[4];
  const float* eW1   = (const float*)d_in[5];
  const float* eb1   = (const float*)d_in[6];
  const float* eW2   = (const float*)d_in[7];
  const float* eb2   = (const float*)d_in[8];
  const float* Tbias = (const float*)d_in[9];
  const float* fW0   = (const float*)d_in[10];
  const float* fb0   = (const float*)d_in[11];
  const float* fW1   = (const float*)d_in[12];
  const float* fb1   = (const float*)d_in[13];
  const float* fW2   = (const float*)d_in[14];
  const float* fb2   = (const float*)d_in[15];
  const float* Ebias = (const float*)d_in[16];
  float* out = (float*)d_out;

  // workspace carve-up (16-byte aligned sections)
  char* ws = (char*)d_ws;
  size_t off = 0;
  _Float16* srcn_h = (_Float16*)(ws + off); off += (size_t)NATOM * MNBR * 2;        // 1 MB
  _Float16* R_h    = (_Float16*)(ws + off); off += (size_t)NATOM * 16 * MNBR * 2;   // 16 MB
  float*    T_f    = (float*)(ws + off);    off += (size_t)NATOM * XDIM * CDIM * 4; // 10 MB
  _Float16* G_h    = (_Float16*)(ws + off); off += (size_t)NATOM * AXISD * CDIM * 2;// 8 MB
  _Float16* W0_t   = (_Float16*)(ws + off); off += (size_t)2 * 1024 * 128 * 2;      // 512 KB
  _Float16* W1_t   = (_Float16*)(ws + off); off += (size_t)2 * 128 * 128 * 2;       // 64 KB

  geom_kernel<<<(NATOM * MNBR) / 256, 256, 0, stream>>>(coord, box, nbrs,
                                                        srcn_h, R_h);
  cvt_kernel<<<1024, 256, 0, stream>>>(fW0, fW1, W0_t, W1_t);
  embed_T_kernel<<<NATOM, 32, 0, stream>>>(srcn_h, R_h, eW0, eb0, eW1, eb1,
                                           eW2, eb2, Tbias, T_f);
  g_kernel<<<NATOM, 32, 0, stream>>>(T_f, G_h);
  zero_out_kernel<<<1, 1, 0, stream>>>(out);
  fit_kernel<<<64, 128, 0, stream>>>(G_h, W0_t, W1_t, fb0, fb1, fW2, fb2,
                                     Ebias, out);
}
